// CVAE_59304908423249
// MI455X (gfx1250) — matrix operand order repaired and checked by reading, not compiled, not run on hardware
//
#include <hip/hip_runtime.h>
#include <stdint.h>

// ---------------- problem dims ----------------
#define Bsz   128
#define Tsz   272
#define Csz   96
#define Hsz   1024
#define Lsz   256
#define MELsz (Tsz * Csz)        // 26112
#define ZDIMs (Lsz + MELsz)      // 26368

typedef __bf16 bf16;
typedef bf16  bf16x16 __attribute__((ext_vector_type(16)));
typedef bf16  bf16x8  __attribute__((ext_vector_type(8)));
typedef float f32x8   __attribute__((ext_vector_type(8)));

#if defined(__has_builtin)
#  if __has_builtin(__builtin_amdgcn_tensor_load_to_lds) && __has_builtin(__builtin_amdgcn_s_wait_tensorcnt)
#    define HAVE_TDM 1
#  endif
#endif
#ifndef HAVE_TDM
#  define HAVE_TDM 0
#endif

#if HAVE_TDM
typedef uint32_t u32x4 __attribute__((ext_vector_type(4)));
typedef int      i32x8 __attribute__((ext_vector_type(8)));
typedef int      i32x4 __attribute__((ext_vector_type(4)));

// 2-D tile DMA: rows x 32 bf16 elements, global row stride = stride_elems,
// packed contiguously into LDS (ld = 32 elements). D# layout per CDNA5 ISA ch.8.
// This toolchain exposes the 6-arg builtin: (g0, g1, g2, g3, g4, cpol).
__device__ __forceinline__ void tdm_load_2d(bf16* lds_dst, const bf16* gsrc,
                                            uint32_t rows, uint32_t stride_elems) {
  const uint32_t row_elems = 32;
  uint64_t ga = (uint64_t)(uintptr_t)gsrc;
  uint32_t la = (uint32_t)(uintptr_t)lds_dst;  // LDS aperture: low 32 bits = LDS offset
  u32x4 g0;
  g0[0] = 1u;                                                  // count=1, user load D#
  g0[1] = la;                                                  // lds_addr (bytes)
  g0[2] = (uint32_t)ga;                                        // global_addr[31:0]
  g0[3] = ((uint32_t)(ga >> 32) & 0x01FFFFFFu) | (2u << 30);   // global_addr[56:32] | type=2
  i32x8 g1;
  g1[0] = (int)(1u << 16);                          // data_size=1 (2 bytes); no multicast/pad
  g1[1] = (int)(row_elems << 16);                   // tensor_dim0[15:0] -> bits 63:48
  g1[2] = (int)((row_elems >> 16) | (rows << 16));  // tensor_dim0[31:16] | tensor_dim1[15:0]
  g1[3] = (int)((rows >> 16) | (row_elems << 16));  // tensor_dim1[31:16] | tile_dim0
  g1[4] = (int)(rows & 0xFFFFu);                    // tile_dim1 (tile_dim2 = 0)
  g1[5] = (int)stride_elems;                        // tensor_dim0_stride[31:0]
  g1[6] = 0;                                        // stride hi16 | dim1_stride lo16
  g1[7] = 0;                                        // dim1_stride hi
  i32x4 gz4 = {0, 0, 0, 0};                         // 2-D tensor: groups 2/3 unused
  i32x8 gz8 = {0, 0, 0, 0, 0, 0, 0, 0};
  __builtin_amdgcn_tensor_load_to_lds(g0, g1, gz4, gz4, gz8, 0);
}
#endif

// ---------------- LDS fragment loads (CDNA5 16-bit WMMA layouts, ld = 32) ----------------
__device__ __forceinline__ bf16x16 lds_a_frag(const bf16* p0, int lane) {
  const bf16* p = p0 + (size_t)(lane & 15) * 32 + ((lane >> 4) << 3);
  bf16x8 lo = *(const bf16x8*)(p);
  bf16x8 hi = *(const bf16x8*)(p + 16);
  bf16x16 v;
#pragma unroll
  for (int i = 0; i < 8; ++i) { v[i] = lo[i]; v[8 + i] = hi[i]; }
  return v;
}
__device__ __forceinline__ bf16x16 lds_b_frag(const bf16* p0, int lane) {
  const bf16* p = p0 + (size_t)(lane & 15) * 32 + ((lane >> 4) << 3);
  bf16x8 lo = *(const bf16x8*)(p);
  bf16x8 hi = *(const bf16x8*)(p + 16);
  bf16x16 v;
#pragma unroll
  for (int i = 0; i < 8; ++i) { v[i] = lo[i]; v[8 + i] = hi[i]; }
  return v;
}

// stage one K-step's tiles into LDS: A 128x32, B nrows x 32
__device__ __forceinline__ void stage_tiles(bf16* dstA, bf16* dstB,
                                            const bf16* __restrict__ A, int lda,
                                            const bf16* __restrict__ Wn0, int ldw,
                                            int k0, int nrows, int tid) {
#if HAVE_TDM
  if ((tid >> 5) == 0) {   // one wave issues both DMAs; tracked in its TENSORcnt
    tdm_load_2d(dstA, A + k0, 128u, (uint32_t)lda);
    tdm_load_2d(dstB, Wn0 + k0, (uint32_t)nrows, (uint32_t)ldw);
  }
#else
  for (int i = tid * 8; i < 128 * 32; i += 256 * 8) {
    int r = i >> 5, c = i & 31;
    *(bf16x8*)(dstA + i) = *(const bf16x8*)(A + (size_t)r * lda + k0 + c);
  }
  for (int i = tid * 8; i < nrows * 32; i += 256 * 8) {
    int r = i >> 5, c = i & 31;
    *(bf16x8*)(dstB + i) = *(const bf16x8*)(Wn0 + (size_t)r * ldw + k0 + c);
  }
#endif
}

// ---------------- GEMM: Y(128, NT*16 per block) = X(128,K) . W(N,K)^T + bias ----------
// block tile 128(M) x 16*NT(N), 8 waves; TDM double-buffered LDS staging; K % 32 == 0.
// NT is compile-time -> branch-free unrolled WMMA loop (EXEC all-ones throughout).
template <int NT>
__global__ __launch_bounds__(256)
void gemm_bf16_wmma(const bf16* __restrict__ A, int lda,
                    const bf16* __restrict__ W, int ldw,
                    const float* __restrict__ bias,
                    float* __restrict__ Co, int ldc,
                    int n_base, int K) {
  __shared__ __align__(64) bf16 smA[2][128 * 32];
  __shared__ __align__(64) bf16 smB[2][64 * 32];
  const int tid  = threadIdx.x;
  const int lane = tid & 31;
  const int wv   = tid >> 5;                 // 0..7 -> M-tile
  const int n0   = n_base + blockIdx.x * 64;
  const int nrows = NT * 16;
  const int nk    = K >> 5;
  const bf16* Wn0 = W + (size_t)n0 * ldw;

  f32x8 acc[NT] = {};

  stage_tiles(smA[0], smB[0], A, lda, Wn0, ldw, 0, nrows, tid);
  for (int ki = 0; ki < nk; ++ki) {
    const int buf = ki & 1;
#if HAVE_TDM
    if (wv == 0) __builtin_amdgcn_s_wait_tensorcnt(0);  // DMA for `buf` complete
#endif
    __syncthreads();
    if (ki + 1 < nk)
      stage_tiles(smA[buf ^ 1], smB[buf ^ 1], A, lda, Wn0, ldw, (ki + 1) << 5, nrows, tid);
    // A fragment reused across the NT N-tiles of this wave
    bf16x16 af = lds_a_frag(&smA[buf][wv * 16 * 32], lane);
#pragma unroll
    for (int nt = 0; nt < NT; ++nt) {
      bf16x16 bfr = lds_b_frag(&smB[buf][nt * 16 * 32], lane);
      acc[nt] = __builtin_amdgcn_wmma_f32_16x16x32_bf16(false, af, false, bfr,
                                                        (short)0, acc[nt], false, false);
    }
    __syncthreads();
  }

  // D layout: lane holds n = (lane&15); VGPR v holds m = v + (lane>=16)*8
  const int mb = wv * 16 + ((lane >> 4) << 3);
#pragma unroll
  for (int nt = 0; nt < NT; ++nt) {
    int n = n0 + nt * 16 + (lane & 15);
    float bv = bias ? bias[n] : 0.0f;
#pragma unroll
    for (int v = 0; v < 8; ++v)
      Co[(size_t)(mb + v) * ldc + n] = acc[nt][v] + bv;
  }
}

// ---------------- GRU elementwise update (gates order r,z,n) ----------------
__global__ __launch_bounds__(256)
void gru_elem(const float* __restrict__ gi, const float* __restrict__ gh,
              float* __restrict__ h32, bf16* __restrict__ hbf,
              bf16* __restrict__ y, int ystride,
              const int* __restrict__ length, int t) {
  int idx = blockIdx.x * blockDim.x + threadIdx.x;
  if (idx >= Bsz * Hsz) return;
  int b = idx >> 10;
  int j = idx & (Hsz - 1);
  size_t g = (size_t)b * (3 * Hsz) + j;
  float r  = gi[g] + gh[g];
  float z  = gi[g + Hsz] + gh[g + Hsz];
  float in = gi[g + 2 * Hsz];
  float hn = gh[g + 2 * Hsz];
  r = 1.0f / (1.0f + __expf(-r));
  z = 1.0f / (1.0f + __expf(-z));
  float n = tanhf(in + r * hn);
  float hprev = h32[idx];
  float hnew  = (1.0f - z) * n + z * hprev;
  bool valid = (length == nullptr) || (t < length[b]);
  float h = valid ? hnew : hprev;
  h32[idx] = h;
  hbf[idx] = (bf16)h;
  if (y) y[(size_t)b * ystride + j] = valid ? (bf16)h : (bf16)0.0f;
}

// ---------------- small helper kernels ----------------
__global__ void cvt_f32_bf16(const float* __restrict__ s, bf16* __restrict__ d, long long n) {
  long long i = (long long)blockIdx.x * blockDim.x + threadIdx.x;
  long long st = (long long)gridDim.x * blockDim.x;
  for (; i < n; i += st) d[i] = (bf16)s[i];
}
__global__ void zero_u32(uint32_t* __restrict__ p, long long n) {
  long long i = (long long)blockIdx.x * blockDim.x + threadIdx.x;
  long long st = (long long)gridDim.x * blockDim.x;
  for (; i < n; i += st) p[i] = 0u;
}
// x (B,T,C) fp32 -> xb (T,B,C) bf16
__global__ void xpose_bf16(const float* __restrict__ x, bf16* __restrict__ xb) {
  int idx = blockIdx.x * blockDim.x + threadIdx.x;
  if (idx >= Tsz * Bsz * Csz) return;
  int c = idx % Csz;
  int r = idx / Csz;
  int b = r % Bsz;
  int t = r / Bsz;
  xb[idx] = (bf16)x[((size_t)b * Tsz + t) * Csz + c];
}
// hid = concat(h0f,h0b,h1f,h1b) -> (B,4H) bf16
__global__ void build_hid(const float* __restrict__ h0f, const float* __restrict__ h0b,
                          const float* __restrict__ h1f, const float* __restrict__ h1b,
                          bf16* __restrict__ hid) {
  int idx = blockIdx.x * blockDim.x + threadIdx.x;
  if (idx >= Bsz * 4 * Hsz) return;
  int b = idx / (4 * Hsz);
  int j = idx % (4 * Hsz);
  int part = j / Hsz, jj = j % Hsz;
  const float* s = (part == 0) ? h0f : (part == 1) ? h0b : (part == 2) ? h1f : h1b;
  hid[idx] = (bf16)s[(size_t)b * Hsz + jj];
}
// z = [eps*exp(0.5*lv)+mu , melody.flat] -> bf16 (B,ZDIM)
__global__ void build_z(const float* __restrict__ eps, const float* __restrict__ mu,
                        const float* __restrict__ lv, const float* __restrict__ mel,
                        bf16* __restrict__ zb) {
  int idx = blockIdx.x * blockDim.x + threadIdx.x;
  if (idx >= Bsz * ZDIMs) return;
  int b = idx / ZDIMs;
  int j = idx % ZDIMs;
  float v;
  if (j < Lsz) {
    size_t o = (size_t)b * Lsz + j;
    v = eps[o] * __expf(0.5f * lv[o]) + mu[o];
  } else {
    v = mel[(size_t)b * MELsz + (j - Lsz)];
  }
  zb[idx] = (bf16)v;
}
// decoder input xt = [init_tok | z_in] -> (B,192) bf16
__global__ void init_dec(const float* __restrict__ tok0, const float* __restrict__ zin,
                         bf16* __restrict__ xt) {
  int idx = blockIdx.x * blockDim.x + threadIdx.x;
  if (idx >= Bsz * 2 * Csz) return;
  int b = idx / (2 * Csz);
  int c = idx % (2 * Csz);
  float v = (c < Csz) ? tok0[c] : zin[(size_t)b * Csz + (c - Csz)];
  xt[idx] = (bf16)v;
}
// per batch row: argmax over C logits -> one-hot bf16 into xt[:, 0:C]
__global__ void argmax_onehot(const float* __restrict__ chd, int ld, bf16* __restrict__ xt) {
  int b = blockIdx.x;
  int lane = threadIdx.x;
  float best = -3.0e38f;
  int bi = 0;
  for (int c = lane; c < Csz; c += 32) {
    float v = chd[(size_t)b * ld + c];
    if (v > best) { best = v; bi = c; }
  }
  for (int off = 16; off > 0; off >>= 1) {
    float ob = __shfl_xor(best, off, 32);
    int   oi = __shfl_xor(bi,   off, 32);
    if (ob > best || (ob == best && oi < bi)) { best = ob; bi = oi; }
  }
  for (int c = lane; c < Csz; c += 32)
    xt[(size_t)b * (2 * Csz) + c] = (bf16)((c == bi) ? 1.0f : 0.0f);
}
// softmax + log_softmax over last dim, one wave per row
__global__ void softmax_logsoftmax(const float* __restrict__ res,
                                   float* __restrict__ sm, float* __restrict__ lp) {
  int row = blockIdx.x;
  int lane = threadIdx.x;
  const float* r = res + (size_t)row * Csz;
  float mx = -3.0e38f;
  for (int c = lane; c < Csz; c += 32) mx = fmaxf(mx, r[c]);
  for (int off = 16; off > 0; off >>= 1) mx = fmaxf(mx, __shfl_xor(mx, off, 32));
  float s = 0.0f;
  for (int c = lane; c < Csz; c += 32) s += __expf(r[c] - mx);
  for (int off = 16; off > 0; off >>= 1) s += __shfl_xor(s, off, 32);
  float inv = 1.0f / s;
  float ls  = __logf(s);
  for (int c = lane; c < Csz; c += 32) {
    float e = r[c] - mx;
    sm[(size_t)row * Csz + c] = __expf(e) * inv;
    lp[(size_t)row * Csz + c] = e - ls;
  }
}

// ---------------- host orchestration ----------------
extern "C" void kernel_launch(void* const* d_in, const int* in_sizes, int n_in,
                              void* d_out, int out_size, void* d_ws, size_t ws_size,
                              hipStream_t stream) {
  (void)in_sizes; (void)n_in; (void)out_size; (void)ws_size;
  const float* in_x   = (const float*)d_in[0];
  const float* melody = (const float*)d_in[1];
  const float* eps    = (const float*)d_in[2];
  const float* Wih[4] = {(const float*)d_in[3],  (const float*)d_in[7],
                         (const float*)d_in[11], (const float*)d_in[15]};
  const float* Whh[4] = {(const float*)d_in[4],  (const float*)d_in[8],
                         (const float*)d_in[12], (const float*)d_in[16]};
  const float* bih[4] = {(const float*)d_in[5],  (const float*)d_in[9],
                         (const float*)d_in[13], (const float*)d_in[17]};
  const float* bhh[4] = {(const float*)d_in[6],  (const float*)d_in[10],
                         (const float*)d_in[14], (const float*)d_in[18]};
  const float* W_mu = (const float*)d_in[19]; const float* b_mu = (const float*)d_in[20];
  const float* W_lv = (const float*)d_in[21]; const float* b_lv = (const float*)d_in[22];
  const float* W_zi = (const float*)d_in[23]; const float* b_zi = (const float*)d_in[24];
  const float* W_zh = (const float*)d_in[25]; const float* b_zh = (const float*)d_in[26];
  const float* tok0 = (const float*)d_in[27];
  const float* dWih = (const float*)d_in[28]; const float* dWhh = (const float*)d_in[29];
  const float* dbih = (const float*)d_in[30]; const float* dbhh = (const float*)d_in[31];
  const float* W_o  = (const float*)d_in[32]; const float* b_o  = (const float*)d_in[33];
  const int*   leng = (const int*)d_in[34];

  const size_t BTC = (size_t)Bsz * Tsz * Csz;
  float* sm = (float*)d_out;
  float* lp = sm + BTC;
  float* mu = sm + 2 * BTC;
  float* lv = mu + (size_t)Bsz * Lsz;

  // workspace sub-allocator
  char* wsb = (char*)d_ws;
  size_t off = 0;
  auto alloc = [&](size_t bytes) -> char* {
    char* p = wsb + off;
    off = (off + bytes + 255) & ~(size_t)255;
    return p;
  };
  auto allocBF = [&](size_t n) -> bf16*  { return (bf16*) alloc(n * 2); };
  auto allocF  = [&](size_t n) -> float* { return (float*)alloc(n * 4); };

  // bf16 weight images (all stay resident in the 192MB L2 across the 272-step loops)
  bf16* wWih0[2] = {allocBF(3 * Hsz * Csz),     allocBF(3 * Hsz * Csz)};       // e0f,e0b
  bf16* wWih1[2] = {allocBF(3 * Hsz * 2 * Hsz), allocBF(3 * Hsz * 2 * Hsz)};   // e1f,e1b
  bf16* wWhh[4]  = {allocBF(3 * Hsz * Hsz), allocBF(3 * Hsz * Hsz),
                    allocBF(3 * Hsz * Hsz), allocBF(3 * Hsz * Hsz)};
  bf16* wWmu  = allocBF((size_t)Lsz * 4 * Hsz);
  bf16* wWlv  = allocBF((size_t)Lsz * 4 * Hsz);
  bf16* wWzi  = allocBF((size_t)Csz * ZDIMs);
  bf16* wWzh  = allocBF((size_t)Hsz * ZDIMs);
  bf16* wdWih = allocBF(3 * Hsz * 2 * Csz);
  bf16* wdWhh = allocBF(3 * Hsz * Hsz);
  bf16* wWo   = allocBF((size_t)Csz * Hsz);
  // activations
  bf16*  xb   = allocBF((size_t)Tsz * Bsz * Csz);
  bf16*  x1b  = allocBF((size_t)Tsz * Bsz * 2 * Hsz);
  float* h32  = allocF((size_t)5 * Bsz * Hsz);   // h0f,h0b,h1f,h1b,hdec (contiguous)
  bf16*  hbf  = allocBF((size_t)5 * Bsz * Hsz);
  float* gi   = allocF((size_t)Bsz * 3 * Hsz);
  float* gh   = allocF((size_t)Bsz * 3 * Hsz);
  bf16*  hid4 = allocBF((size_t)Bsz * 4 * Hsz);
  bf16*  zb   = allocBF((size_t)Bsz * ZDIMs);
  float* zin  = allocF((size_t)Bsz * Csz);
  bf16*  xt   = allocBF((size_t)Bsz * 2 * Csz);
  float* res  = allocF(BTC);

  auto cvt = [&](const float* s, bf16* d, size_t n) {
    int blocks = (int)(((n + 255) / 256) > 16384 ? 16384 : ((n + 255) / 256));
    cvt_f32_bf16<<<blocks, 256, 0, stream>>>(s, d, (long long)n);
  };
  // dispatch: full 64-wide blocks use NT=4; a 16/32/48-wide tail gets its own launch
  auto gemm = [&](const bf16* A, int lda, const bf16* W, int ldw, const float* bias,
                  float* Co, int ldc, int N, int K) {
    int nfull = N / 64;
    int rem   = N % 64;
    if (nfull)
      gemm_bf16_wmma<4><<<dim3(nfull), 256, 0, stream>>>(A, lda, W, ldw, bias, Co, ldc, 0, K);
    if (rem == 16)
      gemm_bf16_wmma<1><<<dim3(1), 256, 0, stream>>>(A, lda, W, ldw, bias, Co, ldc, nfull * 64, K);
    else if (rem == 32)
      gemm_bf16_wmma<2><<<dim3(1), 256, 0, stream>>>(A, lda, W, ldw, bias, Co, ldc, nfull * 64, K);
    else if (rem == 48)
      gemm_bf16_wmma<3><<<dim3(1), 256, 0, stream>>>(A, lda, W, ldw, bias, Co, ldc, nfull * 64, K);
  };

  // 1) weight + input conversions
  cvt(Wih[0], wWih0[0], (size_t)3 * Hsz * Csz);
  cvt(Wih[1], wWih0[1], (size_t)3 * Hsz * Csz);
  cvt(Wih[2], wWih1[0], (size_t)3 * Hsz * 2 * Hsz);
  cvt(Wih[3], wWih1[1], (size_t)3 * Hsz * 2 * Hsz);
  for (int i = 0; i < 4; ++i) cvt(Whh[i], wWhh[i], (size_t)3 * Hsz * Hsz);
  cvt(W_mu, wWmu, (size_t)Lsz * 4 * Hsz);
  cvt(W_lv, wWlv, (size_t)Lsz * 4 * Hsz);
  cvt(W_zi, wWzi, (size_t)Csz * ZDIMs);
  cvt(W_zh, wWzh, (size_t)Hsz * ZDIMs);
  cvt(dWih, wdWih, (size_t)3 * Hsz * 2 * Csz);
  cvt(dWhh, wdWhh, (size_t)3 * Hsz * Hsz);
  cvt(W_o,  wWo,  (size_t)Csz * Hsz);
  xpose_bf16<<<(Tsz * Bsz * Csz + 255) / 256, 256, 0, stream>>>(in_x, xb);

  // 2) zero encoder hidden states (fp32 + bf16 copies)
  zero_u32<<<2048, 256, 0, stream>>>((uint32_t*)h32, (long long)4 * Bsz * Hsz);
  zero_u32<<<1024, 256, 0, stream>>>((uint32_t*)hbf, (long long)2 * Bsz * Hsz);

  float* hs[5]; bf16* hb[5];
  for (int i = 0; i < 5; ++i) { hs[i] = h32 + (size_t)i * Bsz * Hsz; hb[i] = hbf + (size_t)i * Bsz * Hsz; }
  const int eg = (Bsz * Hsz + 255) / 256;

  // 3) encoder layer 0 (fwd + bwd per step), y -> x1b (T,B,2H)
  for (int t = 0; t < Tsz; ++t) {
    gemm(xb + (size_t)t * Bsz * Csz, Csz, wWih0[0], Csz, bih[0], gi, 3 * Hsz, 3 * Hsz, Csz);
    gemm(hb[0], Hsz, wWhh[0], Hsz, bhh[0], gh, 3 * Hsz, 3 * Hsz, Hsz);
    gru_elem<<<eg, 256, 0, stream>>>(gi, gh, hs[0], hb[0],
                                     x1b + (size_t)t * Bsz * 2 * Hsz, 2 * Hsz, leng, t);
    int tb = Tsz - 1 - t;
    gemm(xb + (size_t)tb * Bsz * Csz, Csz, wWih0[1], Csz, bih[1], gi, 3 * Hsz, 3 * Hsz, Csz);
    gemm(hb[1], Hsz, wWhh[1], Hsz, bhh[1], gh, 3 * Hsz, 3 * Hsz, Hsz);
    gru_elem<<<eg, 256, 0, stream>>>(gi, gh, hs[1], hb[1],
                                     x1b + (size_t)tb * Bsz * 2 * Hsz + Hsz, 2 * Hsz, leng, tb);
  }

  // 4) encoder layer 1 (needs full x1b); only final hiddens are consumed
  for (int t = 0; t < Tsz; ++t) {
    gemm(x1b + (size_t)t * Bsz * 2 * Hsz, 2 * Hsz, wWih1[0], 2 * Hsz, bih[2], gi, 3 * Hsz, 3 * Hsz, 2 * Hsz);
    gemm(hb[2], Hsz, wWhh[2], Hsz, bhh[2], gh, 3 * Hsz, 3 * Hsz, Hsz);
    gru_elem<<<eg, 256, 0, stream>>>(gi, gh, hs[2], hb[2], nullptr, 0, leng, t);
    int tb = Tsz - 1 - t;
    gemm(x1b + (size_t)tb * Bsz * 2 * Hsz, 2 * Hsz, wWih1[1], 2 * Hsz, bih[3], gi, 3 * Hsz, 3 * Hsz, 2 * Hsz);
    gemm(hb[3], Hsz, wWhh[3], Hsz, bhh[3], gh, 3 * Hsz, 3 * Hsz, Hsz);
    gru_elem<<<eg, 256, 0, stream>>>(gi, gh, hs[3], hb[3], nullptr, 0, leng, tb);
  }

  // 5) mu / log_var (written straight into d_out)
  build_hid<<<(Bsz * 4 * Hsz + 255) / 256, 256, 0, stream>>>(hs[0], hs[1], hs[2], hs[3], hid4);
  gemm(hid4, 4 * Hsz, wWmu, 4 * Hsz, b_mu, mu, Lsz, Lsz, 4 * Hsz);
  gemm(hid4, 4 * Hsz, wWlv, 4 * Hsz, b_lv, lv, Lsz, Lsz, 4 * Hsz);

  // 6) reparameterize + melody concat; z projections
  build_z<<<(Bsz * ZDIMs + 255) / 256, 256, 0, stream>>>(eps, mu, lv, melody, zb);
  gemm(zb, ZDIMs, wWzi, ZDIMs, b_zi, zin,  Csz, Csz, ZDIMs);
  gemm(zb, ZDIMs, wWzh, ZDIMs, b_zh, hs[4], Hsz, Hsz, ZDIMs);
  cvt(hs[4], hb[4], (size_t)Bsz * Hsz);
  init_dec<<<(Bsz * 2 * Csz + 255) / 256, 256, 0, stream>>>(tok0, zin, xt);

  // 7) autoregressive decoder (argmax one-hot feedback, tfr=0)
  for (int t = 0; t < Tsz; ++t) {
    gemm(xt, 2 * Csz, wdWih, 2 * Csz, dbih, gi, 3 * Hsz, 3 * Hsz, 2 * Csz);
    gemm(hb[4], Hsz, wdWhh, Hsz, dbhh, gh, 3 * Hsz, 3 * Hsz, Hsz);
    gru_elem<<<eg, 256, 0, stream>>>(gi, gh, hs[4], hb[4], nullptr, 0, nullptr, 0);
    gemm(hb[4], Hsz, wWo, Hsz, b_o, res + (size_t)t * Csz, Tsz * Csz, Csz, Hsz);
    argmax_onehot<<<Bsz, 32, 0, stream>>>(res + (size_t)t * Csz, Tsz * Csz, xt);
  }

  // 8) softmax / log_softmax over (B,T,C)
  softmax_logsoftmax<<<Bsz * Tsz, 32, 0, stream>>>(res, sm, lp);
}
